// AgentAttention_37168646979727
// MI455X (gfx1250) — matrix operand order repaired and checked by reading, not compiled, not run on hardware
//
#include <hip/hip_runtime.h>

// ---------------------------------------------------------------------------
// Problem constants
// ---------------------------------------------------------------------------
#define BB      32
#define NN      1024
#define CC      1024
#define HH      16
#define HD      64
#define AA      49
#define IND     3

typedef __bf16 v16bf __attribute__((ext_vector_type(16)));
typedef float  v8f   __attribute__((ext_vector_type(8)));

union Frag16 {          // one WMMA 16-bf16 operand = two 16-byte LDS loads
    v16bf v;
    int4  q[2];
};

// ---------------------------------------------------------------------------
// 1) Input embedding: [B,N,3] @ [3,C] + b  ->  bf16 [B,N,C]
//    Qe = Q@w_vel+b_vel ; Ke = K@w_init+b_init ; Ve = V@w_init+b_init
// ---------------------------------------------------------------------------
__global__ __launch_bounds__(256) void embed_kernel(
    const float* __restrict__ Q, const float* __restrict__ K,
    const float* __restrict__ V,
    const float* __restrict__ w_vel, const float* __restrict__ b_vel,
    const float* __restrict__ w_init, const float* __restrict__ b_init,
    __bf16* __restrict__ Qe, __bf16* __restrict__ Ke, __bf16* __restrict__ Ve)
{
    int idx = blockIdx.x * 256 + threadIdx.x;           // over B*N*C
    if (idx >= BB * NN * CC) return;
    int c  = idx & (CC - 1);
    int bn = idx >> 10;                                  // CC = 1024
    float q0 = Q[bn * 3 + 0], q1 = Q[bn * 3 + 1], q2 = Q[bn * 3 + 2];
    float k0 = K[bn * 3 + 0], k1 = K[bn * 3 + 1], k2 = K[bn * 3 + 2];
    float v0 = V[bn * 3 + 0], v1 = V[bn * 3 + 1], v2 = V[bn * 3 + 2];
    float wv0 = w_vel[c],  wv1 = w_vel[CC + c],  wv2 = w_vel[2 * CC + c];
    float wi0 = w_init[c], wi1 = w_init[CC + c], wi2 = w_init[2 * CC + c];
    Qe[idx] = (__bf16)(q0 * wv0 + q1 * wv1 + q2 * wv2 + b_vel[c]);
    Ke[idx] = (__bf16)(k0 * wi0 + k1 * wi1 + k2 * wi2 + b_init[c]);
    Ve[idx] = (__bf16)(v0 * wi0 + v1 * wi1 + v2 * wi2 + b_init[c]);
}

// ---------------------------------------------------------------------------
// 2) Convert + TRANSPOSE wq, wk, wv (each C*C f32, [K][N] row-major)
//    -> bf16 Wt[sel][N][K].  Makes B-tile staging loads K-contiguous so
//    WMMA B fragments are two ds_load_b128 per lane (same as A fragments).
// ---------------------------------------------------------------------------
__global__ __launch_bounds__(256) void cvt_wT_kernel(
    const float* __restrict__ wq, const float* __restrict__ wk,
    const float* __restrict__ wv, __bf16* __restrict__ out)
{
    int idx = blockIdx.x * 256 + threadIdx.x;           // over 3*C*C
    if (idx >= 3 * CC * CC) return;
    int sel = idx >> 20;                                 // C*C = 1<<20
    int r   = idx & ((CC * CC) - 1);
    int n   = r >> 10;                                   // output row (N)
    int k   = r & (CC - 1);                              // output col (K)
    const float* src = (sel == 0) ? wq : (sel == 1) ? wk : wv;
    out[idx] = (__bf16)src[(size_t)k * CC + n];
}

// ---------------------------------------------------------------------------
// 3) WMMA GEMM: [M=B*N, K=C] (bf16) x Wt[N, K] (bf16, pre-transposed) + bias
//    Block tile 128x128, K-step 32, 8 waves (2x4), wave tile 64x32.
//    Both A and B LDS tiles are K-contiguous per output row/col so every
//    fragment is exactly two 16B ds loads (no scalar inserts, no spills).
// ---------------------------------------------------------------------------
__global__ __launch_bounds__(256) void gemm3_wmma_kernel(
    const __bf16* __restrict__ Qe, const __bf16* __restrict__ Ke,
    const __bf16* __restrict__ Ve,
    const __bf16* __restrict__ WTall,  // [wqT | wkT | wvT], each [N][K] bf16
    const float* __restrict__ bq, const float* __restrict__ bk,
    const float* __restrict__ bv,
    float* __restrict__ Qp, float* __restrict__ Kp, float* __restrict__ Vp)
{
    const int Nd = CC;
    const int Kd = CC;

    const int z = blockIdx.z;
    const __bf16* A    = (z == 0) ? Qe : (z == 1) ? Ke : Ve;
    const __bf16* Wt   = WTall + (size_t)z * CC * CC;
    const float*  bias = (z == 0) ? bq : (z == 1) ? bk : bv;
    float*        Out  = (z == 0) ? Qp : (z == 1) ? Kp : Vp;

    // K-contiguous LDS tiles, padded rows (80B pitch) to spread banks.
    __shared__ __bf16 As[128][40];      // A rows  m0..m0+127, K-slice of 32
    __shared__ __bf16 Bs[128][40];      // W cols  n0..n0+127, K-slice of 32

    const int tid  = threadIdx.x;
    const int m0   = blockIdx.x * 128;
    const int n0   = blockIdx.y * 128;

    const int wave = tid >> 5;           // 0..7
    const int lane = tid & 31;
    const int wm   = wave & 1;           // 2 waves along M (64 rows each)
    const int wn   = wave >> 1;          // 4 waves along N (32 cols each)

    // staging: each thread moves one 8-bf16 (int4) chunk per tile
    const int srow  = tid >> 1;          // 0..127
    const int shalf = (tid & 1) * 16;    // 0 or 16

    v8f acc[4][2];
    for (int mm = 0; mm < 4; ++mm)
        for (int nn = 0; nn < 2; ++nn)
            for (int j = 0; j < 8; ++j) acc[mm][nn][j] = 0.0f;

    const int colq = lane & 15;
    const int kbA  = (lane < 16) ? 0 : 8;    // A K-offset per half-wave
    const int kbB  = (lane < 16) ? 0 : 8;    // B K-offset per half-wave

    const __bf16* Aptr = A  + (size_t)(m0 + srow) * Kd + shalf;
    const __bf16* Bptr = Wt + (size_t)(n0 + srow) * Kd + shalf;

    for (int k0 = 0; k0 < Kd; k0 += 32) {
        // stage tiles (coalesced int4 global loads -> b128 LDS writes)
        int4 av = *(const int4*)(Aptr + k0);
        int4 bw = *(const int4*)(Bptr + k0);
        *(int4*)&As[srow][shalf] = av;
        *(int4*)&Bs[srow][shalf] = bw;
        if (k0 + 32 < Kd) {                  // global_prefetch_b8 next tile
            __builtin_prefetch(Aptr + k0 + 32, 0, 3);
            __builtin_prefetch(Bptr + k0 + 32, 0, 3);
        }
        __syncthreads();

        // B fragments: elems 0..7 = K kbB..kbB+7 ; elems 8..15 = kbB+16..+23
        Frag16 bf[2];
        #pragma unroll
        for (int nn = 0; nn < 2; ++nn) {
            const int cg = wn * 32 + nn * 16 + colq;
            bf[nn].q[0] = *(const int4*)&Bs[cg][kbB];
            bf[nn].q[1] = *(const int4*)&Bs[cg][kbB + 16];
        }

        #pragma unroll
        for (int mm = 0; mm < 4; ++mm) {
            const int rg = wm * 64 + mm * 16 + colq;
            // A fragment: elems 0..7 = K kbA..kbA+7 ; elems 8..15 = kbA+16..+23
            Frag16 af;
            af.q[0] = *(const int4*)&As[rg][kbA];
            af.q[1] = *(const int4*)&As[rg][kbA + 16];
            acc[mm][0] = __builtin_amdgcn_wmma_f32_16x16x32_bf16(
                false, af.v, false, bf[0].v, (short)0, acc[mm][0], false, false);
            acc[mm][1] = __builtin_amdgcn_wmma_f32_16x16x32_bf16(
                false, af.v, false, bf[1].v, (short)0, acc[mm][1], false, false);
        }
        __syncthreads();
    }

    // epilogue: C/D layout -> lane col = lane&15, rows rbase+j (rbase 0 or 8)
    const int rbase = (lane < 16) ? 0 : 8;
    for (int mm = 0; mm < 4; ++mm)
        for (int nn = 0; nn < 2; ++nn)
            #pragma unroll
            for (int j = 0; j < 8; ++j) {
                const int rg = m0 + wm * 64 + mm * 16 + rbase + j;
                const int cg = n0 + wn * 32 + nn * 16 + colq;
                Out[(size_t)rg * Nd + cg] = acc[mm][nn][j] + bias[cg];
            }
}

// ---------------------------------------------------------------------------
// 4) Adaptive average pool: Apool[b,a,c] = mean_{n in [s,e)} Qp[b,n,c]
// ---------------------------------------------------------------------------
__global__ __launch_bounds__(256) void pool_kernel(
    const float* __restrict__ Qp, float* __restrict__ Apool)
{
    int blk = blockIdx.x;                 // b*A + a
    int a = blk % AA, b = blk / AA;
    int s = (a * NN) / AA;
    int e = ((a + 1) * NN + AA - 1) / AA; // ceil
    float inv = 1.0f / (float)(e - s);
    int c = threadIdx.x * 4;
    float4 acc = make_float4(0.f, 0.f, 0.f, 0.f);
    for (int n = s; n < e; ++n) {
        float4 v = *(const float4*)(Qp + ((size_t)b * NN + n) * CC + c);
        acc.x += v.x; acc.y += v.y; acc.z += v.z; acc.w += v.w;
    }
    acc.x *= inv; acc.y *= inv; acc.z *= inv; acc.w *= inv;
    *(float4*)(Apool + ((size_t)b * AA + a) * CC + c) = acc;
}

// ---------------------------------------------------------------------------
// 5) Agent aggregation: per (b,h,a) -> softmax_n(agent . Kh + bias1) @ Vh
// ---------------------------------------------------------------------------
__global__ __launch_bounds__(256) void agent_agg_kernel(
    const float* __restrict__ Apool, const float* __restrict__ Kp,
    const float* __restrict__ Vp, const float* __restrict__ bias1,
    float* __restrict__ AV)
{
    int blk = blockIdx.x;               // (b*H + h)*A + a
    int a  = blk % AA;
    int bh = blk / AA;
    int h  = bh % HH, b = bh / HH;

    __shared__ float ag[HD];
    __shared__ float probs[NN];
    __shared__ float red[256];

    int tid = threadIdx.x;
    if (tid < HD) ag[tid] = Apool[((size_t)b * AA + a) * CC + h * HD + tid];
    __syncthreads();

    float sc[4];
    float lmax = -1e30f;
    #pragma unroll
    for (int k = 0; k < 4; ++k) {
        int n = k * 256 + tid;
        const float* krow = Kp + ((size_t)b * NN + n) * CC + h * HD;
        float s = bias1[a * NN + n];
        #pragma unroll
        for (int d = 0; d < HD; d += 4) {
            float4 kv = *(const float4*)(krow + d);
            s += ag[d] * kv.x + ag[d + 1] * kv.y + ag[d + 2] * kv.z + ag[d + 3] * kv.w;
        }
        sc[k] = s;
        lmax = fmaxf(lmax, s);
    }
    red[tid] = lmax; __syncthreads();
    for (int off = 128; off > 0; off >>= 1) {
        if (tid < off) red[tid] = fmaxf(red[tid], red[tid + off]);
        __syncthreads();
    }
    float gmax = red[0]; __syncthreads();

    float lsum = 0.f;
    #pragma unroll
    for (int k = 0; k < 4; ++k) {
        float e = __expf(sc[k] - gmax);
        probs[k * 256 + tid] = e;
        lsum += e;
    }
    red[tid] = lsum; __syncthreads();
    for (int off = 128; off > 0; off >>= 1) {
        if (tid < off) red[tid] += red[tid + off];
        __syncthreads();
    }
    float inv = 1.0f / red[0];
    __syncthreads();

    // output: d = tid&63, chunk of 256 n's = tid>>6
    int d = tid & 63, ch = tid >> 6;
    float accv = 0.f;
    for (int n = ch * 256; n < ch * 256 + 256; ++n)
        accv += probs[n] * Vp[((size_t)b * NN + n) * CC + h * HD + d];
    red[tid] = accv; __syncthreads();
    if (tid < 64) {
        float o = (red[tid] + red[tid + 64] + red[tid + 128] + red[tid + 192]) * inv;
        AV[(size_t)blk * HD + tid] = o;
    }
}

// ---------------------------------------------------------------------------
// 6) Agent broadcast: per (b,h,n) -> softmax_a(Qh . agent + bias2) @ AV
// ---------------------------------------------------------------------------
__global__ __launch_bounds__(64) void agent_bc_kernel(
    const float* __restrict__ Qp, const float* __restrict__ Apool,
    const float* __restrict__ AV, const float* __restrict__ bias2,
    float* __restrict__ O)
{
    int blk = blockIdx.x;               // (b*H + h)*N + n
    int n  = blk % NN;
    int bh = blk / NN;
    int h  = bh % HH, b = bh / HH;

    __shared__ float qv[HD];
    __shared__ float sc[AA];

    int tid = threadIdx.x;
    qv[tid] = Qp[((size_t)b * NN + n) * CC + h * HD + tid];
    __syncthreads();
    if (tid < AA) {
        const float* ar = Apool + ((size_t)b * AA + tid) * CC + h * HD;
        float s = bias2[n * AA + tid];
        #pragma unroll 8
        for (int d = 0; d < HD; ++d) s += qv[d] * ar[d];
        sc[tid] = s;
    }
    __syncthreads();

    float mx = -1e30f;
    for (int a = 0; a < AA; ++a) mx = fmaxf(mx, sc[a]);
    float sum = 0.f, o = 0.f;
    const float* avb = AV + (size_t)bh * AA * HD;
    for (int a = 0; a < AA; ++a) {
        float p = __expf(sc[a] - mx);
        sum += p;
        o += p * avb[a * HD + tid];
    }
    O[((size_t)b * NN + n) * CC + h * HD + tid] = o / sum;
}

// ---------------------------------------------------------------------------
// 7) Depthwise 3-tap conv (only center column of 3x3 sees data) + residual,
//    final projection C->3, transpose-store as [B, 3, N]
// ---------------------------------------------------------------------------
__global__ __launch_bounds__(128) void fuse_out_kernel(
    const float* __restrict__ O, const float* __restrict__ Vp,
    const float* __restrict__ dwc_w, const float* __restrict__ dwc_b,
    const float* __restrict__ w_proj, const float* __restrict__ b_proj,
    float* __restrict__ out)
{
    int blk = blockIdx.x;                 // b*N + n
    int n = blk % NN, b = blk / NN;
    int tid = threadIdx.x;
    __shared__ float red[3 * 128];

    const size_t base = ((size_t)b * NN + n) * CC;
    float a0 = 0.f, a1 = 0.f, a2 = 0.f;
    for (int c = tid; c < CC; c += 128) {
        float wl = dwc_w[c * 9 + 1];      // kh=0, kw=1
        float wc = dwc_w[c * 9 + 4];      // kh=1, kw=1
        float wr = dwc_w[c * 9 + 7];      // kh=2, kw=1
        float x = O[base + c] + dwc_b[c] + wc * Vp[base + c];
        if (n > 0)      x += wl * Vp[base - CC + c];
        if (n < NN - 1) x += wr * Vp[base + CC + c];
        a0 += x * w_proj[c * 3 + 0];
        a1 += x * w_proj[c * 3 + 1];
        a2 += x * w_proj[c * 3 + 2];
    }
    red[tid] = a0; red[128 + tid] = a1; red[256 + tid] = a2;
    __syncthreads();
    for (int off = 64; off > 0; off >>= 1) {
        if (tid < off) {
            red[tid]       += red[tid + off];
            red[128 + tid] += red[128 + tid + off];
            red[256 + tid] += red[256 + tid + off];
        }
        __syncthreads();
    }
    if (tid == 0) {
        out[((size_t)b * 3 + 0) * NN + n] = red[0]   + b_proj[0];
        out[((size_t)b * 3 + 1) * NN + n] = red[128] + b_proj[1];
        out[((size_t)b * 3 + 2) * NN + n] = red[256] + b_proj[2];
    }
}

// ---------------------------------------------------------------------------
// Launcher
// ---------------------------------------------------------------------------
extern "C" void kernel_launch(void* const* d_in, const int* in_sizes, int n_in,
                              void* d_out, int out_size, void* d_ws, size_t ws_size,
                              hipStream_t stream)
{
    const float* Q      = (const float*)d_in[0];
    const float* K      = (const float*)d_in[1];
    const float* V      = (const float*)d_in[2];
    const float* w_vel  = (const float*)d_in[3];
    const float* b_vel  = (const float*)d_in[4];
    const float* w_init = (const float*)d_in[5];
    const float* b_init = (const float*)d_in[6];
    const float* wq     = (const float*)d_in[7];
    const float* bq     = (const float*)d_in[8];
    const float* wk     = (const float*)d_in[9];
    const float* bk     = (const float*)d_in[10];
    const float* wv     = (const float*)d_in[11];
    const float* bv     = (const float*)d_in[12];
    const float* bias1  = (const float*)d_in[13];
    const float* bias2  = (const float*)d_in[14];
    const float* dwc_w  = (const float*)d_in[15];
    const float* dwc_b  = (const float*)d_in[16];
    const float* w_proj = (const float*)d_in[17];
    const float* b_proj = (const float*)d_in[18];
    float* out = (float*)d_out;

    // workspace carve-up
    char* ws = (char*)d_ws;
    size_t off = 0;
    const size_t sz_bnc_bf = (size_t)BB * NN * CC * sizeof(__bf16);   // 64 MB
    const size_t sz_w_bf   = (size_t)3 * CC * CC * sizeof(__bf16);    //  6 MB
    const size_t sz_bnc_f  = (size_t)BB * NN * CC * sizeof(float);    // 128 MB
    const size_t sz_pool   = (size_t)BB * AA * CC * sizeof(float);
    const size_t sz_av     = (size_t)BB * HH * AA * HD * sizeof(float);

    __bf16* Qe   = (__bf16*)(ws + off); off += sz_bnc_bf;
    __bf16* Ke   = (__bf16*)(ws + off); off += sz_bnc_bf;
    __bf16* Ve   = (__bf16*)(ws + off); off += sz_bnc_bf;
    __bf16* WTall= (__bf16*)(ws + off); off += sz_w_bf;
    float*  Qp   = (float*) (ws + off); off += sz_bnc_f;
    float*  Kp   = (float*) (ws + off); off += sz_bnc_f;
    float*  Vp   = (float*) (ws + off); off += sz_bnc_f;
    float*  Apool= (float*) (ws + off); off += sz_pool;
    float*  AV   = (float*) (ws + off); off += sz_av;
    float*  O    = (float*) (ws + off); off += sz_bnc_f;

    // 1) embed to bf16
    {
        int total = BB * NN * CC;
        embed_kernel<<<(total + 255) / 256, 256, 0, stream>>>(
            Q, K, V, w_vel, b_vel, w_init, b_init, Qe, Ke, Ve);
    }
    // 2) convert + transpose weights
    {
        int total = 3 * CC * CC;
        cvt_wT_kernel<<<(total + 255) / 256, 256, 0, stream>>>(wq, wk, wv, WTall);
    }
    // 3) three big GEMMs via WMMA (z selects wq/wk/wv)
    {
        dim3 grid((BB * NN) / 128, CC / 128, 3);
        gemm3_wmma_kernel<<<grid, 256, 0, stream>>>(
            Qe, Ke, Ve, WTall, bq, bk, bv, Qp, Kp, Vp);
    }
    // 4) pool
    pool_kernel<<<BB * AA, 256, 0, stream>>>(Qp, Apool);
    // 5) agent aggregation
    agent_agg_kernel<<<BB * HH * AA, 256, 0, stream>>>(Apool, Kp, Vp, bias1, AV);
    // 6) agent broadcast
    agent_bc_kernel<<<BB * HH * NN, 64, 0, stream>>>(Qp, Apool, AV, bias2, O);
    // 7) conv + residual + final projection + transpose
    fuse_out_kernel<<<BB * NN, 128, 0, stream>>>(O, Vp, dwc_w, dwc_b, w_proj, b_proj, out);

    (void)in_sizes; (void)n_in; (void)out_size; (void)ws_size;
}
